// MultiHeadAttention_12025908429067
// MI455X (gfx1250) — matrix operand order repaired and checked by reading, not compiled, not run on hardware
//
#include <hip/hip_runtime.h>

#define B_ 4
#define T_ 2048
#define E_ 512
#define H_ 8
#define D_ 64

typedef __attribute__((ext_vector_type(16))) _Float16 v16h;
typedef __attribute__((ext_vector_type(8)))  float    v8f;
typedef __attribute__((ext_vector_type(4)))  unsigned int v4u;
typedef __attribute__((ext_vector_type(8)))  int      v8i;
typedef __attribute__((ext_vector_type(4)))  int      v4i;

union Frag { v16h v; unsigned int u[8]; };
union H2U  { _Float16 h[2]; unsigned int u; };

#if defined(__has_builtin)
#if __has_builtin(__builtin_amdgcn_tensor_load_to_lds) && \
    __has_builtin(__builtin_amdgcn_s_wait_tensorcnt)
#define USE_TDM 1
#endif
#endif
#ifndef USE_TDM
#define USE_TDM 0
#endif

static __device__ inline unsigned int pack2f16(float a, float b) {
  H2U t; t.h[0] = (_Float16)a; t.h[1] = (_Float16)b; return t.u;
}

static __device__ inline v8f wmma_f16(const Frag& a, const Frag& b, v8f c) {
  // D = A(16x32 f16) * B(32x16 f16) + C(16x16 f32)
  return __builtin_amdgcn_wmma_f32_16x16x32_f16(false, a.v, false, b.v,
                                                (short)0, c, false, false);
}

#if USE_TDM
// LDS byte offset of a __shared__ object (generic -> AS(3) -> integer).
static __device__ inline unsigned lds_off(const void* p) {
  return (unsigned)(unsigned long long)
      (__attribute__((address_space(3))) const void*)p;
}

// TDM: async-load a 64x64 f16 tile (row stride 64 elems in memory) into LDS,
// padding each 32-DWORD row with 4 DWORDs -> LDS row stride 72 f16.
// D# packed per CDNA5 ISA "Tensor DMA Descriptor" (CDNA5 ISA).
static __device__ inline void tdm_load_tile64x64_f16(unsigned ldsAddr,
                                                     unsigned long long gaddr,
                                                     unsigned tensorRows) {
  v4u g0;
  g0.x = 1u;                                          // count=1, user mode
  g0.y = ldsAddr;                                     // lds_addr (bytes)
  g0.z = (unsigned)(gaddr & 0xffffffffull);           // global_addr[31:0]
  g0.w = (unsigned)((gaddr >> 32) & 0x01ffffffull)    // global_addr[56:32]
         | (2u << 30);                                // type = 2 ("image")
  v8i g1;
  g1[0] = (int)((1u << 16)      // data_size = 1 -> 2 bytes
              | (1u << 20)      // pad_enable
              | (4u << 22)      // pad_interval: 2<<4 = 32 DWORDs (one row)
              | (3u << 25));    // pad_amount: 3+1 = 4 DWORDs (8 f16)
  g1[1] = (int)(64u << 16);                           // tensor_dim0 = 64 (lo16)
  g1[2] = (int)((tensorRows & 0xffffu) << 16);        // dim0 hi16=0 | dim1 lo16
  g1[3] = (int)((tensorRows >> 16) | (64u << 16));    // dim1 hi16 | tile_dim0=64
  g1[4] = (int)64;                                    // tile_dim1=64, tile_dim2=0
  g1[5] = (int)64;                                    // tensor_dim0_stride = 64
  g1[6] = 0;                                          // stride0 hi | stride1 lo
  g1[7] = 0;                                          // tensor_dim1_stride hi
  v4i z4 = (v4i)0;                                    // groups 2/3 unused (2D)
  v8i z8 = (v8i)0;                                    // extra group (6-arg form)
  __builtin_amdgcn_tensor_load_to_lds(g0, g1, z4, z4, z8, 0);
}
#endif

// ---------------------------------------------------------------------------
// Kernel 1: per-head QKV projection.  C[T,D] = src[T,E] * W[h]^T  (f16 WMMA)
// grid: (T/64, B*H, 3)  block: 128 (4 waves, each owns a 16-row strip)
// ---------------------------------------------------------------------------
__global__ __launch_bounds__(128) void mha_proj_kernel(
    const float* __restrict__ q, const float* __restrict__ k,
    const float* __restrict__ v,
    const float* __restrict__ Wq, const float* __restrict__ Wk,
    const float* __restrict__ Wv,
    _Float16* __restrict__ qh, _Float16* __restrict__ kh,
    _Float16* __restrict__ vh)
{
  const int mt   = blockIdx.x;              // 64-row tile over T
  const int bh   = blockIdx.y;              // b*H + head
  const int b    = bh / H_, head = bh % H_;
  const int sel  = blockIdx.z;

  const float* src = sel == 0 ? q  : (sel == 1 ? k  : v);
  const float* W   = sel == 0 ? Wq : (sel == 1 ? Wk : Wv);
  _Float16*    dst = sel == 0 ? qh : (sel == 1 ? kh : vh);
  const float  scale = sel == 0 ? 0.125f : 1.0f;   // 1/sqrt(D) folded into Q

  __shared__ _Float16 aT[64][40];   // [t_local][e_local] (32 + pad)
  __shared__ _Float16 wT[64][40];   // [d][e_local]

  const int tid  = threadIdx.x;
  const int wave = tid >> 5;
  const int lane = tid & 31;
  const int l16  = lane & 15;
  const int hh   = lane >> 4;       // lane half

  const int t0 = mt * 64;
  const float* srcBase = src + (size_t)b * T_ * E_;
  const float* wBase   = W   + (size_t)head * D_ * E_;

  v8f acc[4] = {};                  // 4 N-subtiles (D = 64)

  for (int e0 = 0; e0 < E_; e0 += 32) {
    for (int i = tid; i < 64 * 32; i += 128) {
      int r = i >> 5, c = i & 31;
      aT[r][c] = (_Float16)(srcBase[(size_t)(t0 + r) * E_ + e0 + c] * scale);
      wT[r][c] = (_Float16)(wBase [(size_t)r        * E_ + e0 + c]);
    }
    __syncthreads();

    // A fragment (M = t strip rows, K = e)
    Frag a;
    const int trow = wave * 16 + l16;
#pragma unroll
    for (int vi = 0; vi < 8; ++vi) {
      int e = 2 * vi + (vi >= 4 ? 8 : 0) + (hh ? 8 : 0);
      a.u[vi] = *(const unsigned int*)&aT[trow][e];
    }
#pragma unroll
    for (int ns = 0; ns < 4; ++ns) {
      Frag bf;
      const int d = ns * 16 + l16;
#pragma unroll
      for (int vi = 0; vi < 8; ++vi) {
        int e = 2 * vi + (vi >= 4 ? 8 : 0) + (hh ? 8 : 0);
        bf.u[vi] = *(const unsigned int*)&wT[d][e];
      }
      acc[ns] = wmma_f16(a, bf, acc[ns]);
    }
    __syncthreads();
  }

  _Float16* dBase = dst + ((size_t)bh * T_) * D_;
#pragma unroll
  for (int ns = 0; ns < 4; ++ns)
#pragma unroll
    for (int g = 0; g < 8; ++g) {
      int t = t0 + wave * 16 + g + (hh ? 8 : 0);
      int d = ns * 16 + l16;
      dBase[(size_t)t * D_ + d] = (_Float16)acc[ns][g];
    }
}

// ---------------------------------------------------------------------------
// Kernel 2: causal flash attention per (b,head).
// Computes S^T = K*Q^T so softmax (over s) is per-lane + one xor16 shuffle,
// then out^T = V^T * P^T.  K tiles stream through the TDM, double-buffered.
// grid: (T/64, B*H), block 128.
// ---------------------------------------------------------------------------
__global__ __launch_bounds__(128) void mha_attn_kernel(
    const _Float16* __restrict__ qh, const _Float16* __restrict__ kh,
    const _Float16* __restrict__ vh, _Float16* __restrict__ aout)
{
  const int qt = blockIdx.x;                // 64-query tile
  const int bh = blockIdx.y;
  const int b  = bh / H_, head = bh % H_;

  __shared__ _Float16 qT[64][72];           // [t_local][d]
  __shared__ _Float16 kT[2][64][72];        // [s_local][d], double-buffered
  __shared__ _Float16 vT[64][72];           // transposed: [d][s_local]

  const int tid  = threadIdx.x;
  const int wave = tid >> 5;
  const int lane = tid & 31;
  const int l16  = lane & 15;
  const int hh   = lane >> 4;

  const _Float16* Qb = qh + (size_t)bh * T_ * D_;
  const _Float16* Kb = kh + (size_t)bh * T_ * D_;
  const _Float16* Vb = vh + (size_t)bh * T_ * D_;

  const int t0blk = qt * 64;

  // ---- load Q tile, build resident Q^T B-fragments (K=d, N=t) ----
  for (int i = tid; i < 2048; i += 128) {
    int r = i >> 5, c = (i & 31) * 2;
    *(unsigned int*)&qT[r][c] =
        *(const unsigned int*)&Qb[(size_t)(t0blk + r) * D_ + c];
  }

#if USE_TDM
  const unsigned long long kbAddr = (unsigned long long)(size_t)(const void*)Kb;
  if (wave == 0)   // prologue: DMA first K tile into buffer 0
    tdm_load_tile64x64_f16(lds_off(&kT[0][0][0]), kbAddr, T_);
#endif
  __syncthreads();

  Frag qf[2];
#pragma unroll
  for (int ks = 0; ks < 2; ++ks)
#pragma unroll
    for (int vi = 0; vi < 8; ++vi) {
      int d = 2 * vi + (vi >= 4 ? 8 : 0) + (hh ? 8 : 0) + 32 * ks;
      qf[ks].u[vi] = *(const unsigned int*)&qT[wave * 16 + l16][d];
    }

  v8f   acc[4] = {};                        // out^T accum: 4 d-subtiles
  float mRow = -3.0e38f, lRow = 0.0f;       // stats for column t = l16

  const int tBase = t0blk + wave * 16;
  const int myT   = tBase + l16;
  const int nTiles = qt + 1;                // causal bound
  int cur = 0;

  for (int s0t = 0; s0t < nTiles; ++s0t) {
    const int s0 = s0t * 64;
    __syncthreads();                        // prior readers of vT / kT done
    for (int i = tid; i < 2048; i += 128) {
      int r = i >> 5, c = (i & 31) * 2;
      H2U t2;
      t2.u = *(const unsigned int*)&Vb[(size_t)(s0 + r) * D_ + c];
      vT[c][r]     = t2.h[0];               // transpose V into LDS
      vT[c + 1][r] = t2.h[1];
#if !USE_TDM
      *(unsigned int*)&kT[cur][r][c] =
          *(const unsigned int*)&Kb[(size_t)(s0 + r) * D_ + c];
#endif
    }
#if USE_TDM
    if (wave == 0) __builtin_amdgcn_s_wait_tensorcnt(0);  // kT[cur] landed
#endif
    __syncthreads();
#if USE_TDM
    if (wave == 0 && s0t + 1 < nTiles)      // overlap next K DMA with compute
      tdm_load_tile64x64_f16(lds_off(&kT[cur ^ 1][0][0]),
                             kbAddr + (unsigned long long)(s0 + 64) * D_ * 2,
                             T_);
#endif
    if (s0t + 1 < nTiles)                   // prefetch next V tile
      __builtin_prefetch(Vb + (size_t)(s0 + 64) * D_ + tid * 32, 0, 0);

    // ---- S^T tiles: M = s (4 subtiles), N = t (this wave's strip) ----
    v8f st[4] = {};
#pragma unroll
    for (int ms = 0; ms < 4; ++ms) {
#pragma unroll
      for (int ks = 0; ks < 2; ++ks) {
        Frag ka;
        const int srow = 16 * ms + l16;
#pragma unroll
        for (int vi = 0; vi < 8; ++vi) {
          int d = 2 * vi + (vi >= 4 ? 8 : 0) + (hh ? 8 : 0) + 32 * ks;
          ka.u[vi] = *(const unsigned int*)&kT[cur][srow][d];
        }
        st[ms] = wmma_f16(ka, qf[ks], st[ms]);
      }
    }

    // ---- causal mask (element: s = s0+16*ms+g+8*hh, t = myT) ----
    if (s0 + 63 > tBase) {
#pragma unroll
      for (int ms = 0; ms < 4; ++ms)
#pragma unroll
        for (int g = 0; g < 8; ++g) {
          int s = s0 + 16 * ms + g + (hh ? 8 : 0);
          if (s > myT) st[ms][g] = -3.0e38f;
        }
    }

    // ---- online softmax over s (per-lane regs + xor16 across halves) ----
    float tmax = st[0][0];
#pragma unroll
    for (int ms = 0; ms < 4; ++ms)
#pragma unroll
      for (int g = 0; g < 8; ++g) tmax = fmaxf(tmax, st[ms][g]);
    tmax = fmaxf(tmax, __shfl_xor(tmax, 16, 32));

    const float mNew = fmaxf(mRow, tmax);
    const float corr = __expf(mRow - mNew);
    float psum = 0.0f;
#pragma unroll
    for (int ms = 0; ms < 4; ++ms)
#pragma unroll
      for (int g = 0; g < 8; ++g) {
        float p = __expf(st[ms][g] - mNew);
        st[ms][g] = p;
        psum += p;
      }
    psum += __shfl_xor(psum, 16, 32);
    lRow = lRow * corr + psum;
    mRow = mNew;
#pragma unroll
    for (int ds = 0; ds < 4; ++ds)
#pragma unroll
      for (int g = 0; g < 8; ++g) acc[ds][g] *= corr;

    // ---- out^T += V^T * P^T : build P^T B-frags from C-layout tiles ----
#pragma unroll
    for (int ks = 0; ks < 2; ++ks) {
      const v8f T0 = st[2 * ks];
      const v8f T1 = st[2 * ks + 1];
      Frag pf;
#pragma unroll
      for (int vi = 0; vi < 4; ++vi) {
        float r0 = __shfl_xor(T1[2 * vi], 16, 32);
        float r1 = __shfl_xor(T1[2 * vi + 1], 16, 32);
        pf.u[vi] = hh ? pack2f16(r0, r1)
                      : pack2f16(T0[2 * vi], T0[2 * vi + 1]);
      }
#pragma unroll
      for (int vi = 4; vi < 8; ++vi) {
        float r0 = __shfl_xor(T0[2 * vi - 8], 16, 32);
        float r1 = __shfl_xor(T0[2 * vi - 7], 16, 32);
        pf.u[vi] = hh ? pack2f16(T1[2 * vi - 8], T1[2 * vi - 7])
                      : pack2f16(r0, r1);
      }
#pragma unroll
      for (int ds = 0; ds < 4; ++ds) {
        Frag va;
        const int drow = 16 * ds + l16;
#pragma unroll
        for (int vi = 0; vi < 8; ++vi) {
          int s = 2 * vi + (hh ? 16 : 0) + 32 * ks;
          va.u[vi] = *(const unsigned int*)&vT[drow][s];
        }
        acc[ds] = wmma_f16(va, pf, acc[ds]);
      }
    }
    cur ^= 1;
  }

  // ---- epilogue: out[t, head, d] = acc^T / l  (f16, [B,T,H*D] layout) ----
  const float inv = 1.0f / lRow;
  _Float16* ob = aout + (size_t)b * T_ * (H_ * D_) + (size_t)head * D_;
  const int t = tBase + l16;
#pragma unroll
  for (int ds = 0; ds < 4; ++ds)
#pragma unroll
    for (int g = 0; g < 8; ++g) {
      int d = 16 * ds + g + (hh ? 8 : 0);
      ob[(size_t)t * (H_ * D_) + d] = (_Float16)(acc[ds][g] * inv);
    }
}

// ---------------------------------------------------------------------------
// Kernel 3: output projection  out[B*T,E] = aout[B*T,HD] * Wo^T + bo  (f32 out)
// grid: (B*T/64, E/64), block 128.
// ---------------------------------------------------------------------------
__global__ __launch_bounds__(128) void mha_outproj_kernel(
    const _Float16* __restrict__ aout, const float* __restrict__ Wo,
    const float* __restrict__ bo, float* __restrict__ out)
{
  const int mt = blockIdx.x;                // row tile over B*T
  const int nt = blockIdx.y;                // col tile over E

  __shared__ _Float16 aT[64][40];
  __shared__ _Float16 wT[64][40];

  const int tid  = threadIdx.x;
  const int wave = tid >> 5;
  const int lane = tid & 31;
  const int l16  = lane & 15;
  const int hh   = lane >> 4;

  const int m0 = mt * 64, n0 = nt * 64;
  const int K  = H_ * D_;                   // 512

  v8f acc[4] = {};

  for (int e0 = 0; e0 < K; e0 += 32) {
    for (int i = tid; i < 1024; i += 128) {
      int r = i >> 4, c = (i & 15) * 2;
      *(unsigned int*)&aT[r][c] =
          *(const unsigned int*)&aout[(size_t)(m0 + r) * K + e0 + c];
    }
    for (int i = tid; i < 2048; i += 128) {
      int r = i >> 5, c = i & 31;
      wT[r][c] = (_Float16)Wo[(size_t)(n0 + r) * K + e0 + c];
    }
    __syncthreads();

    Frag a;
    const int mrow = wave * 16 + l16;
#pragma unroll
    for (int vi = 0; vi < 8; ++vi) {
      int e = 2 * vi + (vi >= 4 ? 8 : 0) + (hh ? 8 : 0);
      a.u[vi] = *(const unsigned int*)&aT[mrow][e];
    }
#pragma unroll
    for (int ns = 0; ns < 4; ++ns) {
      Frag bf;
      const int d = ns * 16 + l16;
#pragma unroll
      for (int vi = 0; vi < 8; ++vi) {
        int e = 2 * vi + (vi >= 4 ? 8 : 0) + (hh ? 8 : 0);
        bf.u[vi] = *(const unsigned int*)&wT[d][e];
      }
      acc[ns] = wmma_f16(a, bf, acc[ns]);
    }
    __syncthreads();
  }

#pragma unroll
  for (int ns = 0; ns < 4; ++ns)
#pragma unroll
    for (int g = 0; g < 8; ++g) {
      int row = m0 + wave * 16 + g + (hh ? 8 : 0);
      int col = n0 + ns * 16 + l16;
      out[(size_t)row * E_ + col] = acc[ns][g] + bo[col];
    }
}

// ---------------------------------------------------------------------------
extern "C" void kernel_launch(void* const* d_in, const int* in_sizes, int n_in,
                              void* d_out, int out_size, void* d_ws,
                              size_t ws_size, hipStream_t stream) {
  (void)in_sizes; (void)n_in; (void)out_size; (void)ws_size;
  // setup_inputs order: x, k, q, v, Wq, Wk, Wv, Wo, bo  (all f32)
  const float* k_  = (const float*)d_in[1];
  const float* q_  = (const float*)d_in[2];
  const float* v_  = (const float*)d_in[3];
  const float* Wq  = (const float*)d_in[4];
  const float* Wk  = (const float*)d_in[5];
  const float* Wv  = (const float*)d_in[6];
  const float* Wo  = (const float*)d_in[7];
  const float* bo  = (const float*)d_in[8];
  float* out = (float*)d_out;

  const size_t N = (size_t)B_ * H_ * T_ * D_;   // 4,194,304 f16 per buffer
  _Float16* qh = (_Float16*)d_ws;
  _Float16* kh = qh + N;
  _Float16* vh = kh + N;
  _Float16* ao = vh + N;                        // [B, T, H*D] f16

  dim3 g1(T_ / 64, B_ * H_, 3);
  mha_proj_kernel<<<g1, 128, 0, stream>>>(q_, k_, v_, Wq, Wk, Wv, qh, kh, vh);

  dim3 g2(T_ / 64, B_ * H_);
  mha_attn_kernel<<<g2, 128, 0, stream>>>(qh, kh, vh, ao);

  dim3 g3((B_ * T_) / 64, E_ / 64);
  mha_outproj_kernel<<<g3, 128, 0, stream>>>(ao, Wo, bo, out);
}
